// SimpleAttention_24876450579215
// MI455X (gfx1250) — matrix operand order repaired and checked by reading, not compiled, not run on hardware
//
#include <hip/hip_runtime.h>
#include <hip/hip_bf16.h>

#define T_DIM 2048
#define B_DIM 32
#define F_DIM 1024
#define H_DIM 1024
#define MASK_PENALTY (-99999.0f)

#define NT_CH 32                         // T-chunks per batch (blocks per b)
#define T_PER_BLOCK (T_DIM / NT_CH)      // 64
#define T_PER_WAVE  (T_PER_BLOCK / 8)    // 8

typedef __attribute__((ext_vector_type(16))) _Float16 v16h;
typedef __attribute__((ext_vector_type(8)))  float    v8f;

// ---------------------------------------------------------------------------
// Wave32 WMMA 16x16x32 f16 tile: D = A(16xK) * B(Kx16), f32 accumulate.
// Layouts used by this code (contiguous k, same order for A and B):
//   A 16x32: lane m=lane&15; VGPR v: K=2v+half + (lane>=16)*16
//   B 32x16: lane n=lane&15; VGPR v: K=2v+half + (lane>=16)*16
//   C/D:     VGPR r: M=r+(lane>=16)*8, N=lane&15
// ---------------------------------------------------------------------------
__device__ __forceinline__ v8f wmma_gemm_tile(const float* __restrict__ A, int ldAm, int ldAk,
                                              const float* __restrict__ B, int ldBk, int ldBn,
                                              int K)
{
    const int lane  = threadIdx.x & 31;
    const int lhalf = lane >> 4;
    const int lm    = lane & 15;
    v8f acc = {};
    for (int k0 = 0; k0 < K; k0 += 32) {
        v16h av, bv;
#pragma unroll
        for (int i = 0; i < 16; ++i) {
            const int vg = i >> 1, hi = i & 1;
            const int ka = (vg << 1) + hi + (lhalf << 4);
            av[i] = (_Float16)A[lm * ldAm + (k0 + ka) * ldAk];
            const int kb = (vg << 1) + hi + (lhalf << 4);
            bv[i] = (_Float16)B[(k0 + kb) * ldBk + lm * ldBn];
        }
        acc = __builtin_amdgcn_wmma_f32_16x16x32_f16(false, av, false, bv,
                                                     (short)0, acc, false, false);
    }
    return acc;
}

// ---------------------------------------------------------------------------
// K1: qflat_r[(b*2+c)*H + h] = (hidden @ Wq^T + bq)[b, 2h+c]
// ---------------------------------------------------------------------------
__global__ void k_qflat(const float* __restrict__ hidden, const float* __restrict__ Wq,
                        const float* __restrict__ bq, float* __restrict__ qflat_r)
{
    const int wave  = (blockIdx.x * blockDim.x + threadIdx.x) >> 5;
    const int tileM = wave >> 7;
    const int tileN = wave & 127;
    const float* A = hidden + (size_t)tileM * 16 * F_DIM;
    const float* B = Wq     + (size_t)tileN * 16 * F_DIM;
    v8f acc = wmma_gemm_tile(A, F_DIM, 1, B, 1, F_DIM, F_DIM);
    const int lane = threadIdx.x & 31;
#pragma unroll
    for (int r = 0; r < 8; ++r) {
        const int m = r + ((lane >> 4) << 3);
        const int n = lane & 15;
        const int b = tileM * 16 + m;
        const int j = tileN * 16 + n;
        const float v = acc[r] + bq[j];
        const int c = j & 1, h = j >> 1;
        qflat_r[(size_t)(b * 2 + c) * H_DIM + h] = v;
    }
}

// ---------------------------------------------------------------------------
// K2: weff[row*F + f] = sum_h qflat_r[row*H+h] * Wk[h*F+f],  row = b*2+c
// ---------------------------------------------------------------------------
__global__ void k_weff(const float* __restrict__ qflat_r, const float* __restrict__ Wk,
                       float* __restrict__ weff)
{
    const int wave  = (blockIdx.x * blockDim.x + threadIdx.x) >> 5;
    const int tileM = wave >> 6;
    const int tileN = wave & 63;
    const float* A = qflat_r + (size_t)tileM * 16 * H_DIM;
    const float* B = Wk      + (size_t)tileN * 16;
    v8f acc = wmma_gemm_tile(A, H_DIM, 1, B, F_DIM, 1, H_DIM);
    const int lane = threadIdx.x & 31;
#pragma unroll
    for (int r = 0; r < 8; ++r) {
        const int m = r + ((lane >> 4) << 3);
        const int n = lane & 15;
        weff[(size_t)(tileM * 16 + m) * F_DIM + tileN * 16 + n] = acc[r];
    }
}

// K2b: qbias[row] = sum_h bk[h] * qflat_r[row*H+h]
__global__ void k_qbias(const float* __restrict__ qflat_r, const float* __restrict__ bk,
                        float* __restrict__ qbias)
{
    const int wave = (blockIdx.x * blockDim.x + threadIdx.x) >> 5;
    const int lane = threadIdx.x & 31;
    const float* row = qflat_r + (size_t)wave * H_DIM;
    float s = 0.f;
    for (int h = lane; h < H_DIM; h += 32) s = fmaf(bk[h], row[h], s);
    for (int off = 16; off; off >>= 1) s += __shfl_xor(s, off, 32);
    if (lane == 0) qbias[wave] = s;
}

// ---------------------------------------------------------------------------
// K3 (fused, single pass over features):
// For each (b, T-chunk) block: per wave, stream feature rows once, compute
// both channel scores (store raw), and maintain an online-softmax weighted
// feature accumulator (flash style). Waves combine in LDS -> one partial
// (ctx_part[F], m, S) per block.
// ---------------------------------------------------------------------------
__global__ void k_flash(const float* __restrict__ feat, const float* __restrict__ weff,
                        const float* __restrict__ qbias, const float* __restrict__ mask,
                        float* __restrict__ scores, float* __restrict__ part_ctx,
                        float* __restrict__ part_max, float* __restrict__ part_sum)
{
    __shared__ float4 sh_acc[8][F_DIM / 4];   // 8 waves x 1024 floats = 32 KB
    __shared__ float  sh_m[8], sh_s[8];

    const int b    = blockIdx.x >> 5;         // 0..31
    const int tc   = blockIdx.x & 31;         // 0..31
    const int wv   = threadIdx.x >> 5;        // 0..7
    const int lane = threadIdx.x & 31;
    const int tBase = tc * T_PER_BLOCK + wv * T_PER_WAVE;

    // per-lane slice of the two effective query vectors (stays in VGPRs)
    const float4* w0 = (const float4*)(weff + (size_t)(b * 2 + 0) * F_DIM);
    const float4* w1 = (const float4*)(weff + (size_t)(b * 2 + 1) * F_DIM);
    float4 u[8], v[8];
#pragma unroll
    for (int j = 0; j < 8; ++j) { u[j] = w0[j * 32 + lane]; v[j] = w1[j * 32 + lane]; }
    const float qb0 = qbias[b * 2 + 0], qb1 = qbias[b * 2 + 1];

    float4 acc[8];
#pragma unroll
    for (int j = 0; j < 8; ++j) acc[j] = make_float4(0.f, 0.f, 0.f, 0.f);
    float m = -3.0e38f, S = 0.f;

    for (int it = 0; it < T_PER_WAVE; ++it) {
        const int t = tBase + it;
        const size_t row = (size_t)t * B_DIM + b;
        const float4* fr = (const float4*)(feat + row * F_DIM);
        if (it + 1 < T_PER_WAVE)   // per-lane prefetch of next row (4KB covered)
            __builtin_prefetch(feat + (row + B_DIM) * F_DIM + lane * 32, 0, 3);
        float4 x[8];
#pragma unroll
        for (int j = 0; j < 8; ++j) x[j] = fr[j * 32 + lane];
        float s0 = 0.f, s1 = 0.f;
#pragma unroll
        for (int j = 0; j < 8; ++j) {
            s0 = fmaf(x[j].x, u[j].x, fmaf(x[j].y, u[j].y, fmaf(x[j].z, u[j].z, fmaf(x[j].w, u[j].w, s0))));
            s1 = fmaf(x[j].x, v[j].x, fmaf(x[j].y, v[j].y, fmaf(x[j].z, v[j].z, fmaf(x[j].w, v[j].w, s1))));
        }
        for (int off = 16; off; off >>= 1) {
            s0 += __shfl_xor(s0, off, 32);
            s1 += __shfl_xor(s1, off, 32);
        }
        const float pen = mask[row] * MASK_PENALTY;
        s0 += qb0 + pen;
        s1 += qb1 + pen;
        if (lane == 0) { scores[row * 2 + 0] = s0; scores[row * 2 + 1] = s1; }
        // online softmax update (s0 identical across lanes after butterfly)
        const float mn    = fmaxf(m, s0);
        const float scale = __expf(m - mn);
        const float w     = __expf(s0 - mn);
        S = S * scale + w;
#pragma unroll
        for (int j = 0; j < 8; ++j) {
            acc[j].x = fmaf(x[j].x, w, acc[j].x * scale);
            acc[j].y = fmaf(x[j].y, w, acc[j].y * scale);
            acc[j].z = fmaf(x[j].z, w, acc[j].z * scale);
            acc[j].w = fmaf(x[j].w, w, acc[j].w * scale);
        }
        m = mn;
    }

    // ---- block combine across 8 waves ----
    if (lane == 0) { sh_m[wv] = m; sh_s[wv] = S; }
    __syncthreads();
    float mb = sh_m[0];
#pragma unroll
    for (int i = 1; i < 8; ++i) mb = fmaxf(mb, sh_m[i]);
    float Sb = 0.f;
#pragma unroll
    for (int i = 0; i < 8; ++i) Sb += sh_s[i] * __expf(sh_m[i] - mb);
    const float ws = __expf(m - mb);
#pragma unroll
    for (int j = 0; j < 8; ++j) {
        float4 a = acc[j];
        a.x *= ws; a.y *= ws; a.z *= ws; a.w *= ws;
        sh_acc[wv][j * 32 + lane] = a;
    }
    __syncthreads();
    // thread tid reduces its float4 slot over the 8 waves
    float4 r = sh_acc[0][threadIdx.x];
#pragma unroll
    for (int wvi = 1; wvi < 8; ++wvi) {
        const float4 a = sh_acc[wvi][threadIdx.x];
        r.x += a.x; r.y += a.y; r.z += a.z; r.w += a.w;
    }
    float4* pc4 = (float4*)(part_ctx + (size_t)blockIdx.x * F_DIM);
    pc4[threadIdx.x] = r;
    if (threadIdx.x == 0) { part_max[blockIdx.x] = mb; part_sum[blockIdx.x] = Sb; }
}

// ---------------------------------------------------------------------------
// K4: exact softmax over T per (b,c) from stored scores -> output distributions
// d_out layout: [summary 32768][dist 65536][copy 65536]
// ---------------------------------------------------------------------------
__global__ void k_softmax(const float* __restrict__ scores, float* __restrict__ dout)
{
    __shared__ float red[256];
    const int b = blockIdx.x >> 1, c = blockIdx.x & 1;
    const int tid = threadIdx.x;
    float mx = -1e30f;
    for (int t = tid; t < T_DIM; t += 256)
        mx = fmaxf(mx, scores[((size_t)t * B_DIM + b) * 2 + c]);
    red[tid] = mx; __syncthreads();
    for (int s = 128; s; s >>= 1) { if (tid < s) red[tid] = fmaxf(red[tid], red[tid + s]); __syncthreads(); }
    mx = red[0]; __syncthreads();
    float sm = 0.f;
    for (int t = tid; t < T_DIM; t += 256)
        sm += __expf(scores[((size_t)t * B_DIM + b) * 2 + c] - mx);
    red[tid] = sm; __syncthreads();
    for (int s = 128; s; s >>= 1) { if (tid < s) red[tid] += red[tid + s]; __syncthreads(); }
    const float inv = 1.0f / red[0];
    float* out = dout + B_DIM * H_DIM + (size_t)c * (T_DIM * B_DIM);
    for (int t = tid; t < T_DIM; t += 256)
        out[(size_t)t * B_DIM + b] = __expf(scores[((size_t)t * B_DIM + b) * 2 + c] - mx) * inv;
}

// K5: exact merge of the 32 chunk (m,S) pairs per batch -> mx[b], S[b]
__global__ void k_flash_combine(const float* __restrict__ part_max,
                                const float* __restrict__ part_sum,
                                float* __restrict__ mxS)
{
    const int wave = (blockIdx.x * blockDim.x + threadIdx.x) >> 5;  // = b (0..31)
    const int lane = threadIdx.x & 31;                              // = tc
    const float pm = part_max[wave * NT_CH + lane];
    float m = pm;
    for (int off = 16; off; off >>= 1) m = fmaxf(m, __shfl_xor(m, off, 32));
    float s = part_sum[wave * NT_CH + lane] * __expf(pm - m);
    for (int off = 16; off; off >>= 1) s += __shfl_xor(s, off, 32);
    if (lane == 0) { mxS[wave] = m; mxS[32 + wave] = s; }
}

// K6: ctx[b,f] = (1/S[b]) * sum_tc part_ctx[b,tc,f] * exp(m_tc - mx[b])
__global__ void k_ctx_scale_reduce(const float* __restrict__ part_ctx,
                                   const float* __restrict__ part_max,
                                   const float* __restrict__ mxS,
                                   float* __restrict__ ctx)
{
    const int i = blockIdx.x * blockDim.x + threadIdx.x;  // 0..B*F-1
    const int b = i >> 10;
    const int f = i & (F_DIM - 1);
    const float mx = mxS[b];
    const float inv = 1.0f / mxS[32 + b];
    float s = 0.f;
    for (int tc = 0; tc < NT_CH; ++tc) {
        const int blk = b * NT_CH + tc;
        s = fmaf(part_ctx[(size_t)blk * F_DIM + f], __expf(part_max[blk] - mx), s);
    }
    ctx[i] = s * inv;
}

// ---------------------------------------------------------------------------
// K7: summary[b*H+h] = sum_f Wm[h,f]*ctx[b,f] + bm[h]   (WMMA)
// ---------------------------------------------------------------------------
__global__ void k_summary(const float* __restrict__ ctx, const float* __restrict__ Wm,
                          const float* __restrict__ bm, float* __restrict__ dout)
{
    const int wave  = (blockIdx.x * blockDim.x + threadIdx.x) >> 5;
    const int tileM = wave >> 6;
    const int tileN = wave & 63;
    const float* A = ctx + (size_t)tileM * 16 * F_DIM;
    const float* B = Wm  + (size_t)tileN * 16 * F_DIM;
    v8f acc = wmma_gemm_tile(A, F_DIM, 1, B, 1, F_DIM, F_DIM);
    const int lane = threadIdx.x & 31;
#pragma unroll
    for (int r = 0; r < 8; ++r) {
        const int m = r + ((lane >> 4) << 3);
        const int n = lane & 15;
        const int b = tileM * 16 + m;
        const int h = tileN * 16 + n;
        dout[(size_t)b * H_DIM + h] = acc[r] + bm[h];
    }
}

// ---------------------------------------------------------------------------
extern "C" void kernel_launch(void* const* d_in, const int* in_sizes, int n_in,
                              void* d_out, int out_size, void* d_ws, size_t ws_size,
                              hipStream_t stream)
{
    (void)in_sizes; (void)n_in; (void)out_size; (void)ws_size;
    const float* features = (const float*)d_in[0];
    const float* hidden   = (const float*)d_in[1];
    const float* mask     = (const float*)d_in[2];
    const float* Wk       = (const float*)d_in[3];
    const float* bk       = (const float*)d_in[4];
    const float* Wq       = (const float*)d_in[5];
    const float* bq       = (const float*)d_in[6];
    const float* Wm       = (const float*)d_in[7];
    const float* bm       = (const float*)d_in[8];
    float* out = (float*)d_out;

    // workspace layout (floats): ~5.4 MB total
    float* ws       = (float*)d_ws;
    float* qflat_r  = ws;                                       // 64*1024
    float* weff     = qflat_r + 64 * H_DIM;                     // 64*1024
    float* qbias    = weff    + 64 * F_DIM;                     // 64
    float* scores   = qbias   + 64;                             // T*B*2 = 131072
    float* part_ctx = scores  + (size_t)T_DIM * B_DIM * 2;      // 1024*1024
    float* part_max = part_ctx + (size_t)B_DIM * NT_CH * F_DIM; // 1024
    float* part_sum = part_max + B_DIM * NT_CH;                 // 1024
    float* mxS      = part_sum + B_DIM * NT_CH;                 // 64
    float* ctx      = mxS + 64;                                 // B*F = 32768

    k_qflat           <<<32,   256, 0, stream>>>(hidden, Wq, bq, qflat_r);
    k_weff            <<<32,   256, 0, stream>>>(qflat_r, Wk, weff);
    k_qbias           <<<8,    256, 0, stream>>>(qflat_r, bk, qbias);
    k_flash           <<<1024, 256, 0, stream>>>(features, weff, qbias, mask,
                                                 scores, part_ctx, part_max, part_sum);
    k_softmax         <<<64,   256, 0, stream>>>(scores, out);
    k_flash_combine   <<<4,    256, 0, stream>>>(part_max, part_sum, mxS);
    k_ctx_scale_reduce<<<128,  256, 0, stream>>>(part_ctx, part_max, mxS, ctx);
    k_summary         <<<16,   256, 0, stream>>>(ctx, Wm, bm, out);
}
